// DNN_22557168239122
// MI455X (gfx1250) — matrix operand order repaired and checked by reading, not compiled, not run on hardware
//
#include <hip/hip_runtime.h>
#include <hip/hip_bf16.h>
#include <hip/hip_fp16.h>

// ---------------------------------------------------------------------------
// VQ-VAE forward on MI455X (gfx1250): every conv/deconv = implicit-im2col GEMM
// executed with v_wmma_f32_16x16x32_f16.  f16 activations (compute-bound
// problem: ~410 GFLOP vs ~0.3 GB traffic), f32 accumulate + f32 epilogues.
// This revision software-pipelines the K-loop: ping-pong LDS buffers, global
// fetch issued 2 stages ahead, LDS fill 1 stage ahead, ONE barrier per K-step.
// ---------------------------------------------------------------------------

typedef _Float16 v8h  __attribute__((ext_vector_type(8)));
typedef _Float16 v16h __attribute__((ext_vector_type(16)));
typedef float    v8f  __attribute__((ext_vector_type(8)));
typedef float    v4f  __attribute__((ext_vector_type(4)));

#define BB 2048
#define LL 100
#define DD 64

enum { IN_F32 = 0, IN_F16 = 1, IN_CB = 2 };

__device__ __forceinline__ v16h cat8(v8h lo, v8h hi) {
  return __builtin_shufflevector(lo, hi, 0,1,2,3,4,5,6,7,8,9,10,11,12,13,14,15);
}

// ---------------------------------------------------------------------------
// Implicit-GEMM conv kernel.  Tile: 128(M) x 64(N), BK=32, 8 waves, each wave
// computes a 32x32 subtile (2x2 fragments) -> 4 WMMAs per K-step.
//   A (M x K): im2col rows of the input image (f32 / f16 / codebook-indirect)
//   B (K x N): conv weights [2,2,Cin,Cout] flattened to [K,Cout] (or user_book
//              rows when WTRANS, for the codebook-distance GEMM).
// A is stored in LDS in natural K order (k at position k) so the per-lane
// WMMA A fragment (contiguous k = 16*half + s, same as B) is one contiguous 32B read.
// ---------------------------------------------------------------------------
template<int HIN,int WIN,int CIN,int HOUT,int WOUT,int COUT,int PAD,int KHW,
         int INMODE,bool WTRANS,bool HASBIAS,bool BNRELU,bool OUTF32,bool ACCSQ>
__global__ __launch_bounds__(256)
void conv_gemm(const void* __restrict__ in,
               const int*  __restrict__ code_idx,
               const float* __restrict__ codebook,
               const float* __restrict__ wgt,
               const float* __restrict__ bias,
               const float* __restrict__ gamma,
               const float* __restrict__ bnbeta,
               void* __restrict__ out,
               float* __restrict__ accum)
{
  constexpr int KTOT   = KHW*KHW*CIN;
  constexpr int KSTEPS = KTOT/32;        // 8..64, always even
  constexpr int LDK    = 40;             // 80B row stride: aligned, bank-spread
  constexpr int ASZ    = 128*LDK;
  constexpr int BSZ    = 64*LDK;

  __shared__ _Float16 As[2*ASZ] __attribute__((aligned(16)));   // ping-pong
  __shared__ _Float16 Bs[2*BSZ] __attribute__((aligned(16)));
  __shared__ float red[256];

  const int tid  = threadIdx.x;
  const int lane = tid & 31;
  const int wave = tid >> 5;
  const int wm   = wave & 3;             // 4 waves along M
  const int wn   = wave >> 2;            // 2 waves along N
  const int lm   = lane & 15;
  const int hiH  = lane >> 4;            // lane half (ISA fragment layout)

  const int bm = blockIdx.x;
  const int bn = blockIdx.y;

  // A-loader: 2 threads per tile row, 16 channels each.
  const int arow = tid >> 1;
  const int ah   = tid & 1;
  const int gma  = bm*128 + arow;
  const int ow   = gma % WOUT;
  const int tmp  = gma / WOUT;
  const int oh   = tmp % HOUT;
  const int bi   = tmp / HOUT;

  v8f acc00 = {0.f,0.f,0.f,0.f,0.f,0.f,0.f,0.f};
  v8f acc01 = acc00, acc10 = acc00, acc11 = acc00;

  // ---- stage fetch: global -> registers (issued 2 K-steps ahead) ----------
  auto fetchA = [&](int ks, v8h& c0, v8h& c1) {
    const int kbase = ks*32;
    const int blk = kbase / CIN;               // Cin>=64 -> constant per step
    const int dy  = (KHW==2) ? (blk >> 1) : 0;
    const int dx  = (KHW==2) ? (blk & 1)  : 0;
    const int cib = (kbase % CIN) + ah*16;
    const int ih  = oh + dy - PAD;
    const int iw  = ow + dx - PAD;
    const bool inb = (PAD==0) || (ih>=0 && ih<HIN && iw>=0 && iw<WIN);
    c0 = (v8h){0,0,0,0,0,0,0,0};
    c1 = (v8h){0,0,0,0,0,0,0,0};
    if (inb) {
      if constexpr (INMODE == IN_F16) {
        const _Float16* p = (const _Float16*)in +
            ((size_t)(bi*HIN + ih)*WIN + iw)*CIN + cib;
        c0 = *(const v8h*)p;
        c1 = *(const v8h*)(p + 8);
      } else {
        const float* p;
        if constexpr (INMODE == IN_CB) {       // straight-through quantized
          const int code = code_idx[(bi*HIN + ih)*WIN + iw];
          p = codebook + (size_t)code*CIN + cib;
        } else {
          p = (const float*)in + ((size_t)(bi*HIN + ih)*WIN + iw)*CIN + cib;
        }
        const v4f f0 = *(const v4f*)(p);
        const v4f f1 = *(const v4f*)(p+4);
        const v4f f2 = *(const v4f*)(p+8);
        const v4f f3 = *(const v4f*)(p+12);
        #pragma unroll
        for (int j=0;j<4;++j) {
          c0[j]   = (_Float16)f0[j];
          c0[4+j] = (_Float16)f1[j];
          c1[j]   = (_Float16)f2[j];
          c1[4+j] = (_Float16)f3[j];
        }
      }
    }
  };

  v8h rbt = {0,0,0,0,0,0,0,0};                 // WTRANS staging (user_book)
  v4f rb0 = {0.f,0.f,0.f,0.f}, rb1 = rb0;      // conv-weight staging
  auto fetchB = [&](int ks) {
    const int kbase = ks*32;
    if constexpr (WTRANS) {                    // user_book: [Cout][K]
      const int nb = tid & 63;
      const int k8 = (tid >> 6) * 8;
      const float* wp = wgt + (size_t)(bn*64 + nb)*KTOT + kbase + k8;
      v8h w8;
      #pragma unroll
      for (int j=0;j<8;++j) w8[j] = (_Float16)wp[j];
      rbt = w8;
    } else {                                   // conv weights: [K][Cout]
      const int kc = tid & 31;
      const int n0 = (tid >> 5) * 8;
      const float* wp = wgt + (size_t)(kbase + kc)*COUT + bn*64 + n0;
      rb0 = *(const v4f*)wp;
      rb1 = *(const v4f*)(wp + 4);
    }
  };

  // ---- stage store: registers -> LDS buffer p (issued 1 K-step ahead) -----
  auto stageStore = [&](int p, v8h c0, v8h c1) {
    _Float16* ab = As + p*ASZ;
    // natural-K store: src k 0-31 -> position k
    *(v8h*)&ab[arow*LDK + (ah ? 16 : 0 )] = c0;
    *(v8h*)&ab[arow*LDK + (ah ? 24 : 8 )] = c1;
    _Float16* bb = Bs + p*BSZ;                 // stored [col][k]
    if constexpr (WTRANS) {
      const int nb = tid & 63;
      const int k8 = (tid >> 6) * 8;
      *(v8h*)&bb[nb*LDK + k8] = rbt;
    } else {
      const int kc = tid & 31;
      const int n0 = (tid >> 5) * 8;
      #pragma unroll
      for (int j=0;j<4;++j) {
        bb[(n0+j  )*LDK + kc] = (_Float16)rb0[j];
        bb[(n0+4+j)*LDK + kc] = (_Float16)rb1[j];
      }
    }
  };

  // ---- compute stage: LDS buffer p -> 4 WMMAs -----------------------------
  auto compute = [&](int p) {
    const _Float16* ab = As + p*ASZ;
    const _Float16* bb = Bs + p*BSZ;
    const int aoff = hiH ? 16 : 0;
    const _Float16* a0p = &ab[(wm*32      + lm)*LDK + aoff];
    const _Float16* a1p = &ab[(wm*32 + 16 + lm)*LDK + aoff];
    const _Float16* b0p = &bb[(wn*32      + lm)*LDK + aoff];
    const _Float16* b1p = &bb[(wn*32 + 16 + lm)*LDK + aoff];
    v16h a0 = cat8(*(const v8h*)a0p, *(const v8h*)(a0p+8));
    v16h a1 = cat8(*(const v8h*)a1p, *(const v8h*)(a1p+8));
    v16h b0 = cat8(*(const v8h*)b0p, *(const v8h*)(b0p+8));
    v16h b1 = cat8(*(const v8h*)b1p, *(const v8h*)(b1p+8));
    acc00 = __builtin_amdgcn_wmma_f32_16x16x32_f16(false,a0,false,b0,(short)0,acc00,false,false);
    acc01 = __builtin_amdgcn_wmma_f32_16x16x32_f16(false,a0,false,b1,(short)0,acc01,false,false);
    acc10 = __builtin_amdgcn_wmma_f32_16x16x32_f16(false,a1,false,b0,(short)0,acc10,false,false);
    acc11 = __builtin_amdgcn_wmma_f32_16x16x32_f16(false,a1,false,b1,(short)0,acc11,false,false);
  };

  // ---- software-pipelined main loop: ONE barrier per K-step ---------------
  v8h ra0, ra1;
  fetchA(0, ra0, ra1); fetchB(0);
  stageStore(0, ra0, ra1);
  fetchA(1, ra0, ra1); fetchB(1);              // KSTEPS >= 8 always

  #pragma unroll 2
  for (int ks = 0; ks < KSTEPS; ++ks) {
    __syncthreads();   // buf[ks&1] ready for all; buf[1-(ks&1)] reads retired
    if (ks + 1 < KSTEPS) {
      stageStore((ks + 1) & 1, ra0, ra1);
      if (ks + 2 < KSTEPS) { fetchA(ks + 2, ra0, ra1); fetchB(ks + 2); }
    }
    compute(ks & 1);
  }

  // Epilogue: +bias, inference BN (gamma*rsqrt(1+eps), beta), ReLU, store.
  const float bnmul = rsqrtf(1.0f + 1e-3f);
  float sqacc = 0.f;
  auto store_tile = [&](const v8f& A, int fm, int fn) {
    const int gn = bn*64 + wn*32 + fn*16 + lm;
    float bv = 0.f, sc = 1.f, bt = 0.f;
    if constexpr (HASBIAS) bv = bias[gn];
    if constexpr (BNRELU) { sc = gamma[gn]*bnmul; bt = bnbeta[gn]; }
    const int gm0 = bm*128 + wm*32 + fm*16 + hiH*8;   // D-matrix row mapping
    #pragma unroll
    for (int r=0;r<8;++r) {
      float v = A[r] + bv;
      if constexpr (BNRELU) v = fmaxf(v*sc + bt, 0.f);
      if constexpr (ACCSQ) sqacc += v*v;
      const size_t oidx = (size_t)(gm0 + r)*COUT + gn;
      if constexpr (OUTF32) ((float*)out)[oidx] = v;
      else ((_Float16*)out)[oidx] = (_Float16)v;
    }
  };
  store_tile(acc00,0,0); store_tile(acc01,0,1);
  store_tile(acc10,1,0); store_tile(acc11,1,1);

  if constexpr (ACCSQ) {                  // sum of x_encode^2 for the VQ loss
    red[tid] = sqacc;
    __syncthreads();
    for (int s=128; s>0; s>>=1) { if (tid<s) red[tid]+=red[tid+s]; __syncthreads(); }
    if (tid==0) atomicAdd(accum, red[0]);
  }
}

// --- zero accumulators + precompute codebook norms -------------------------
__global__ void prep_kernel(const float* __restrict__ ub, float* cnorm, float* accum) {
  const int c = threadIdx.x;              // 64 threads
  if (c == 0) { accum[0] = 0.f; accum[1] = 0.f; }
  float s = 0.f;
  for (int k = 0; k < 512; ++k) { const float v = ub[c*512 + k]; s += v*v; }
  cnorm[c] = s;
}

// --- masked embedding gather ------------------------------------------------
__global__ void gather_kernel(const int* __restrict__ ids, const int* __restrict__ masks,
                              const float* __restrict__ table, float* __restrict__ hist) {
  const int gid = blockIdx.x*256 + threadIdx.x;       // BB*LL*16 threads exactly
  const int row = gid >> 4, v = gid & 15;
  const float m = (masks[row] >= 1) ? 1.f : 0.f;
  v4f e = ((const v4f*)(table + (size_t)ids[row]*DD))[v];
  e *= m;
  ((v4f*)(hist + (size_t)row*DD))[v] = e;
}

// --- argmin over 64 codes; dist = ||c||^2 - 2 x.c  (||x||^2 folded in conv3)
__global__ void argmin_kernel(const float* __restrict__ S, const float* __restrict__ cnorm,
                              int* __restrict__ idx, float* __restrict__ accum) {
  __shared__ float scn[64];
  __shared__ float red[256];
  const int tid = threadIdx.x;
  if (tid < 64) scn[tid] = cnorm[tid];
  __syncthreads();
  const int r = blockIdx.x*256 + tid;                 // M divisible by 256
  const float* sr = S + (size_t)r*64;
  float best = scn[0] - 2.f*sr[0]; int bi = 0;
  for (int c = 1; c < 64; ++c) {
    const float d = scn[c] - 2.f*sr[c];
    if (d < best) { best = d; bi = c; }
  }
  idx[r] = bi;
  red[tid] = best;
  __syncthreads();
  for (int s=128; s>0; s>>=1) { if (tid<s) red[tid]+=red[tid+s]; __syncthreads(); }
  if (tid == 0) atomicAdd(accum, red[0]);
}

// --- per-batch masked mean of vq_x + reconstruction SSE ---------------------
__global__ void final_reduce(const int* __restrict__ masks, const float* __restrict__ hist,
                             const float* __restrict__ vqx, float* __restrict__ vqmean,
                             float* __restrict__ accum) {
  __shared__ float sm[256];
  __shared__ float msh;
  const int b = blockIdx.x, tid = threadIdx.x;
  float mv = 0.f;
  if (tid < LL) mv = (masks[b*LL + tid] >= 1) ? 1.f : 0.f;
  sm[tid] = mv; __syncthreads();
  for (int s=128; s>0; s>>=1) { if (tid<s) sm[tid]+=sm[tid+s]; __syncthreads(); }
  if (tid == 0) msh = sm[0];
  __syncthreads();
  const float msum = msh;
  const int d = tid & 63, part = tid >> 6;
  float s = 0.f, sq = 0.f;
  for (int l = part; l < LL; l += 4) {
    const size_t o = (size_t)b*LL*DD + (size_t)l*DD + d;
    const float v = vqx[o], h = hist[o];
    s += v;
    const float df = v - h; sq += df*df;
  }
  sm[tid] = sq; __syncthreads();
  for (int st=128; st>0; st>>=1) { if (tid<st) sm[tid]+=sm[tid+st]; __syncthreads(); }
  if (tid == 0) atomicAdd(accum + 1, sm[0]);
  __syncthreads();
  sm[tid] = s; __syncthreads();
  if (tid < 64) {
    const float tot = sm[tid] + sm[tid+64] + sm[tid+128] + sm[tid+192];
    vqmean[(size_t)b*DD + tid] = tot / msum;
  }
}

__global__ void finalize_loss(const float* __restrict__ accum, float* __restrict__ loss) {
  // mse(hist,vq) + 1.25 * mse(quantized, x_encode)
  loss[0] = accum[1] * (1.0f/13107200.0f) + 1.25f * accum[0] * (1.0f/51380224.0f);
}

// ---------------------------------------------------------------------------
extern "C" void kernel_launch(void* const* d_in, const int* in_sizes, int n_in,
                              void* d_out, int out_size, void* d_ws, size_t ws_size,
                              hipStream_t stream)
{
  const int*   ids   = (const int*)  d_in[0];
  const int*   masks = (const int*)  d_in[1];
  const float* table = (const float*)d_in[2];
  const float* ubook = (const float*)d_in[3];
  const float* e_w1 = (const float*)d_in[4];
  const float* e_b1 = (const float*)d_in[5];
  const float* e_g1 = (const float*)d_in[6];
  const float* e_be1= (const float*)d_in[7];
  const float* e_w2 = (const float*)d_in[8];
  const float* e_b2 = (const float*)d_in[9];
  const float* e_g2 = (const float*)d_in[10];
  const float* e_be2= (const float*)d_in[11];
  const float* e_w3 = (const float*)d_in[12];
  const float* e_b3 = (const float*)d_in[13];
  const float* d_w1 = (const float*)d_in[14];
  const float* d_b1 = (const float*)d_in[15];
  const float* d_g1 = (const float*)d_in[16];
  const float* d_be1= (const float*)d_in[17];
  const float* d_w2 = (const float*)d_in[18];
  const float* d_b2 = (const float*)d_in[19];
  const float* d_g2 = (const float*)d_in[20];
  const float* d_be2= (const float*)d_in[21];
  const float* d_w3 = (const float*)d_in[22];
  const float* d_b3 = (const float*)d_in[23];

  float* out    = (float*)d_out;          // [vq_mean | vq_x | loss]
  float* vqmean = out;
  float* vqx    = out + (size_t)BB*DD;
  float* loss   = out + (size_t)BB*DD + (size_t)BB*LL*DD;

  char* ws = (char*)d_ws;
  constexpr size_t OFF_ACC  = 0;
  constexpr size_t OFF_CN   = 256;
  constexpr size_t OFF_IDX  = 512;                                  // 100352 ints
  constexpr size_t OFF_HIST = 401920;                               // f32 [B,100,64]
  constexpr size_t OFF_ACT1 = OFF_HIST + (size_t)BB*LL*DD*4;        // f16 [B,9,9,128]
  constexpr size_t OFF_ACT2 = OFF_ACT1 + (size_t)BB*81*128*2;       // f16 [B,8,8,256]
  constexpr size_t OFF_XENC = OFF_ACT2 + (size_t)BB*64*256*2;       // f16 [B,7,7,512]
  constexpr size_t OFF_S    = OFF_XENC + (size_t)BB*49*512*2;       // f32 [B*49,64]
  constexpr size_t WS_NEED  = OFF_S    + (size_t)BB*49*64*4;        // ~291 MB
  if (ws_size < WS_NEED) return;

  float*    accum = (float*)(ws + OFF_ACC);
  float*    cnorm = (float*)(ws + OFF_CN);
  int*      idxb  = (int*)  (ws + OFF_IDX);
  float*    hist  = (float*)(ws + OFF_HIST);
  _Float16* act1  = (_Float16*)(ws + OFF_ACT1);
  _Float16* act2  = (_Float16*)(ws + OFF_ACT2);
  _Float16* xenc  = (_Float16*)(ws + OFF_XENC);
  float*    Sbuf  = (float*)(ws + OFF_S);
  _Float16* dec1  = (_Float16*)(ws + OFF_XENC);   // reuse: xenc dead after Sbuf
  _Float16* dec2  = (_Float16*)(ws + OFF_ACT1);   // reuse: act1 dead after conv2

  prep_kernel<<<1, 64, 0, stream>>>(ubook, cnorm, accum);
  gather_kernel<<<12800, 256, 0, stream>>>(ids, masks, table, hist);

  // encoder (VALID convs)
  conv_gemm<10,10, 64, 9, 9,128,0,2,IN_F32,false,true ,true ,false,false>
    <<<dim3(1296,2), 256, 0, stream>>>(hist, nullptr, nullptr, e_w1, e_b1, e_g1, e_be1, act1, accum);
  conv_gemm< 9, 9,128, 8, 8,256,0,2,IN_F16,false,true ,true ,false,false>
    <<<dim3(1024,4), 256, 0, stream>>>(act1, nullptr, nullptr, e_w2, e_b2, e_g2, e_be2, act2, accum);
  conv_gemm< 8, 8,256, 7, 7,512,0,2,IN_F16,false,true ,false,false,true >
    <<<dim3(784, 8), 256, 0, stream>>>(act2, nullptr, nullptr, e_w3, e_b3, nullptr, nullptr, xenc, accum);

  // codebook distance GEMM (x_encode . user_book^T), then argmin
  conv_gemm< 7, 7,512, 7, 7, 64,0,1,IN_F16,true ,false,false,true ,false>
    <<<dim3(784, 1), 256, 0, stream>>>(xenc, nullptr, nullptr, ubook, nullptr, nullptr, nullptr, Sbuf, accum);
  argmin_kernel<<<392, 256, 0, stream>>>(Sbuf, cnorm, idxb, accum);

  // decoder (conv_transpose stride 1 == pad-1 correlation, unflipped kernel)
  conv_gemm< 7, 7,512, 8, 8,256,1,2,IN_CB ,false,true ,true ,false,false>
    <<<dim3(1024,4), 256, 0, stream>>>(nullptr, idxb, ubook, d_w1, d_b1, d_g1, d_be1, dec1, accum);
  conv_gemm< 8, 8,256, 9, 9,128,1,2,IN_F16,false,true ,true ,false,false>
    <<<dim3(1296,2), 256, 0, stream>>>(dec1, nullptr, nullptr, d_w2, d_b2, d_g2, d_be2, dec2, accum);
  conv_gemm< 9, 9,128,10,10, 64,1,2,IN_F16,false,true ,false,true ,false>
    <<<dim3(1600,1), 256, 0, stream>>>(dec2, nullptr, nullptr, d_w3, d_b3, nullptr, nullptr, vqx, accum);

  final_reduce<<<BB, 256, 0, stream>>>(masks, hist, vqx, vqmean, accum);
  finalize_loss<<<1, 1, 0, stream>>>(accum, loss);
}
